// ParallelAttentionFeed_27779848470589
// MI455X (gfx1250) — matrix operand order repaired and checked by reading, not compiled, not run on hardware
//
#include <hip/hip_runtime.h>
#include <math.h>

typedef __attribute__((ext_vector_type(16))) __bf16 v16bf;
typedef __attribute__((ext_vector_type(8)))  float  v8f;

union V16 {                  // 32-byte WMMA 16-bit operand built from two 16B chunks
    uint4  q[2];
    v16bf  v;
};

__device__ inline unsigned short f2bf(float f) {
    unsigned u = __builtin_bit_cast(unsigned, f);
    u += 0x7FFFu + ((u >> 16) & 1u);          // round-to-nearest-even
    return (unsigned short)(u >> 16);
}

__device__ inline v8f v8f_zero() {
    v8f z = {0.f, 0.f, 0.f, 0.f, 0.f, 0.f, 0.f, 0.f};
    return z;
}

// CDNA5 async global->LDS copy (16B per lane), tracked by ASYNCcnt.
__device__ inline void async_b128(unsigned lds_addr, const unsigned short* gaddr) {
    asm volatile("global_load_async_to_lds_b128 %0, %1, off"
                 :: "v"(lds_addr), "v"(gaddr) : "memory");
}
__device__ inline void wait_async0() {
    asm volatile("s_wait_asynccnt 0x0" ::: "memory");
}
__device__ inline unsigned lds_off(const void* p) {
    return (unsigned)(size_t)p;   // generic LDS pointer: addr[31:0] = LDS offset
}

// ---------------------------------------------------------------- fp32 -> bf16
__global__ void cvt_f32_bf16(const float* __restrict__ in,
                             unsigned short* __restrict__ out, int n) {
    int i = (blockIdx.x * blockDim.x + threadIdx.x) * 4;
    if (i + 3 < n) {
        float4 f = *(const float4*)(in + i);
        out[i]     = f2bf(f.x);
        out[i + 1] = f2bf(f.y);
        out[i + 2] = f2bf(f.z);
        out[i + 3] = f2bf(f.w);
    }
}

// ---------------------------------------------- fp32 [K][N] -> bf16 [N][K] (weights)
__global__ __launch_bounds__(256) void cvt_transpose_bf16(
    const float* __restrict__ in, unsigned short* __restrict__ out, int K, int N) {
    __shared__ unsigned short tile[32][33];
    const int tx = threadIdx.x, ty = threadIdx.y;         // block (32, 8)
    const int bn = blockIdx.x * 32, bk = blockIdx.y * 32;
#pragma unroll
    for (int j = 0; j < 4; ++j)
        tile[ty + j * 8][tx] = f2bf(in[(size_t)(bk + ty + j * 8) * N + bn + tx]);
    __syncthreads();
#pragma unroll
    for (int j = 0; j < 4; ++j)
        out[(size_t)(bn + ty + j * 8) * K + bk + tx] = tile[tx][ty + j * 8];
}

// ---------------------------------------------------------------- WMMA GEMM
// C[M,N] = A[M,K] (bf16 row-major) @ B (bf16, pre-transposed BT[N][K]), f32 accum.
// Block tile 128x256, K-step 32, double-buffered LDS fed by async global->LDS.
// 8 waves: 2(M) x 4(N); each wave 64x64 = 4x4 WMMA tiles.
// MODE 0: outF = C            MODE 1: outF = C + bias
// MODE 2: outF += C + bias    MODE 3: outB = bf16(gelu(C + bias))
#define BM 128
#define BN 256
#define BK 32

template <int MODE>
__global__ __launch_bounds__(256) void gemm_bf16(
    const unsigned short* __restrict__ A, const unsigned short* __restrict__ BT,
    const float* __restrict__ bias, float* __restrict__ outF,
    unsigned short* __restrict__ outB, int M, int N, int K) {
    __shared__ unsigned short As[2][BM][BK];   // [m][k], k contiguous
    __shared__ unsigned short Bs[2][BN][BK];   // [n][k], k contiguous

    const int t = threadIdx.x;
    const int lane = t & 31, wid = t >> 5;
    const int L = lane & 15, kh = lane >> 4;
    const int waveM = wid & 1, waveN = wid >> 1;     // 2 x 4 waves
    const int bm0 = blockIdx.y * BM, bn0 = blockIdx.x * BN;

    const int arow = t >> 1, acol = (t & 1) * 16;    // A: 16 elems/thread x2 chunks
    const unsigned short* gA = A + (size_t)(bm0 + arow) * K + acol;
    const unsigned short* gB = BT + (size_t)(bn0 + t) * K;   // B: 32 elems/thread

    v8f acc[4][4];
#pragma unroll
    for (int i = 0; i < 4; ++i)
#pragma unroll
        for (int j = 0; j < 4; ++j) acc[i][j] = v8f_zero();

    auto issue = [&](int k0, int buf) {
        async_b128(lds_off(&As[buf][arow][acol]),     gA + k0);
        async_b128(lds_off(&As[buf][arow][acol + 8]), gA + k0 + 8);
#pragma unroll
        for (int j = 0; j < 4; ++j)
            async_b128(lds_off(&Bs[buf][t][j * 8]), gB + k0 + j * 8);
    };

    issue(0, 0);
    const int nsteps = K / BK;
    for (int s = 0; s < nsteps; ++s) {
        const int buf = s & 1;
        wait_async0();
        __syncthreads();                        // tile s resident for all waves
        if (s + 1 < nsteps) issue((s + 1) * BK, buf ^ 1);

        V16 a[4], b[4];
#pragma unroll
        for (int mt = 0; mt < 4; ++mt) {
            const uint4* p = (const uint4*)&As[buf][waveM * 64 + mt * 16 + L][0];
            a[mt].q[0] = p[kh];            // K = kh*8 + 0..7
            a[mt].q[1] = p[2 + kh];        // K = 16 + kh*8 + 0..7
        }
#pragma unroll
        for (int nt = 0; nt < 4; ++nt) {
            const uint4* p = (const uint4*)&Bs[buf][waveN * 64 + nt * 16 + L][0];
            b[nt].q[0] = p[kh];            // K = kh*8 + 0..7
            b[nt].q[1] = p[2 + kh];        // K = 16 + kh*8 + 0..7
        }
#pragma unroll
        for (int mt = 0; mt < 4; ++mt)
#pragma unroll
            for (int nt = 0; nt < 4; ++nt)
                acc[mt][nt] = __builtin_amdgcn_wmma_f32_16x16x32_bf16(
                    false, a[mt].v, false, b[nt].v, (short)0, acc[mt][nt], false, false);
        __syncthreads();                        // tile s fully consumed
    }

#pragma unroll
    for (int mt = 0; mt < 4; ++mt) {
#pragma unroll
        for (int nt = 0; nt < 4; ++nt) {
            const int gc  = bn0 + waveN * 64 + nt * 16 + L;
            const int gr0 = bm0 + waveM * 64 + mt * 16 + 8 * kh;
#pragma unroll
            for (int r = 0; r < 8; ++r) {
                size_t off = (size_t)(gr0 + r) * N + gc;
                float v = acc[mt][nt][r];
                if (MODE == 0) {
                    outF[off] = v;
                } else if (MODE == 1) {
                    outF[off] = v + bias[gc];
                } else if (MODE == 2) {
                    outF[off] += v + bias[gc];
                } else {
                    float g = v + bias[gc];
                    g = 0.5f * g * (1.0f + erff(g * 0.70710678f));
                    outB[off] = f2bf(g);
                }
            }
        }
    }
}

// ---------------------------------------------------------------- QK layernorm
// One wave per (b,h,n). LN(q), LN(k) over d=64 -> bf16; v -> bf16.
__global__ __launch_bounds__(32) void qk_ln(const float* __restrict__ qkv,
                                            unsigned short* __restrict__ qb,
                                            unsigned short* __restrict__ kb,
                                            unsigned short* __restrict__ vb) {
    const int idx = blockIdx.x;         // bh*2048 + n
    const int bh = idx >> 11, n = idx & 2047;
    const int b = bh >> 4, h = bh & 15;
    const int t = threadIdx.x;
    const size_t src = ((size_t)b * 2048 + n) * 3072;
    const size_t dst = ((size_t)bh * 2048 + n) * 64;
#pragma unroll
    for (int p = 0; p < 2; ++p) {
        const float* s = qkv + src + p * 1024 + h * 64;
        float x0 = s[t], x1 = s[t + 32];
        float sum = x0 + x1, sq = x0 * x0 + x1 * x1;
#pragma unroll
        for (int o = 16; o; o >>= 1) {
            sum += __shfl_xor(sum, o, 32);
            sq  += __shfl_xor(sq, o, 32);
        }
        float mu = sum * (1.0f / 64.0f);
        float rs = rsqrtf(sq * (1.0f / 64.0f) - mu * mu + 1e-5f);
        unsigned short* d = (p ? kb : qb) + dst;
        d[t]      = f2bf((x0 - mu) * rs);
        d[t + 32] = f2bf((x1 - mu) * rs);
    }
    const float* s = qkv + src + 2048 + h * 64;
    vb[dst + t]      = f2bf(s[t]);
    vb[dst + t + 32] = f2bf(s[t + 32]);
}

// ---------------------------------------------------------------- flash attention
// grid (b*h, n/64), 128 threads (4 waves); each wave owns 16 query rows.
__global__ __launch_bounds__(128) void attn(
    const unsigned short* __restrict__ Q, const unsigned short* __restrict__ Kg,
    const unsigned short* __restrict__ Vg, unsigned short* __restrict__ O) {
    __shared__ unsigned short Ks[32][64];      // [key][d]
    __shared__ unsigned short Vt[64][32];      // [d][key] (transposed for B operand)
    __shared__ unsigned short Pw[4][16][32];   // per-wave P re-layout buffer

    const int t = threadIdx.x, lane = t & 31, wid = t >> 5;
    const int L = lane & 15, kh = lane >> 4;
    const int bh = blockIdx.x, b = bh >> 4, h = bh & 15;
    const int q0 = blockIdx.y * 64 + wid * 16;
    const size_t qrow = (size_t)bh * 2048 + q0;

    // Preload the wave's q tile (16x64) as two A operands (K chunks of 32).
    V16 qa[2];
    {
        const uint4* qp = (const uint4*)(Q + (qrow + L) * 64);
#pragma unroll
        for (int c = 0; c < 2; ++c) {
            qa[c].q[0] = qp[c * 4 + kh];
            qa[c].q[1] = qp[c * 4 + 2 + kh];
        }
    }

    float m[8], l[8];
    v8f o[4];
#pragma unroll
    for (int r = 0; r < 8; ++r) { m[r] = -3.0e38f; l[r] = 0.0f; }
#pragma unroll
    for (int d = 0; d < 4; ++d) o[d] = v8f_zero();

    const int key = t >> 2, dc = (t & 3) * 16;
    const float scale = 0.03125f;   // dim^-0.5 = 1/sqrt(1024)

    for (int k0 = 0; k0 < 2048; k0 += 32) {
        const size_t kvrow = ((size_t)bh * 2048 + k0 + key) * 64 + dc;
        // K chunk: straight copy via async global->LDS
        async_b128(lds_off(&Ks[key][dc]),     Kg + kvrow);
        async_b128(lds_off(&Ks[key][dc + 8]), Kg + kvrow + 8);
        // V chunk: transposed into LDS manually
        const uint4* vp = (const uint4*)(Vg + kvrow);
        uint4 v0 = vp[0], v1 = vp[1];
        unsigned short tmp[16];
        *(uint4*)tmp       = v0;
        *(uint4*)(tmp + 8) = v1;
#pragma unroll
        for (int i = 0; i < 16; ++i) Vt[dc + i][key] = tmp[i];
        wait_async0();
        __syncthreads();

        // S = q @ k^T : 16 x 32 (two 16x16 N-tiles, two K=32 steps over d=64)
        v8f s[2];
#pragma unroll
        for (int nt = 0; nt < 2; ++nt) {
            s[nt] = v8f_zero();
            const uint4* p = (const uint4*)&Ks[nt * 16 + L][0];
#pragma unroll
            for (int c = 0; c < 2; ++c) {
                V16 bb;
                bb.q[0] = p[c * 4 + kh];
                bb.q[1] = p[c * 4 + 2 + kh];
                s[nt] = __builtin_amdgcn_wmma_f32_16x16x32_bf16(
                    false, qa[c].v, false, bb.v, (short)0, s[nt], false, false);
            }
        }

        // online softmax; row M lives in one 16-lane half -> width-16 shuffles
#pragma unroll
        for (int r = 0; r < 8; ++r) {
            float s0 = s[0][r] * scale, s1 = s[1][r] * scale;
            float mx = fmaxf(s0, s1);
#pragma unroll
            for (int off = 8; off; off >>= 1) mx = fmaxf(mx, __shfl_xor(mx, off, 16));
            float mnew  = fmaxf(m[r], mx);
            float alpha = __expf(m[r] - mnew);
            float p0 = __expf(s0 - mnew), p1 = __expf(s1 - mnew);
            float rs = p0 + p1;
#pragma unroll
            for (int off = 8; off; off >>= 1) rs += __shfl_xor(rs, off, 16);
            l[r] = l[r] * alpha + rs;
            m[r] = mnew;
#pragma unroll
            for (int d = 0; d < 4; ++d) o[d][r] *= alpha;
            Pw[wid][r + 8 * kh][L]      = f2bf(p0);
            Pw[wid][r + 8 * kh][L + 16] = f2bf(p1);
        }
        __syncthreads();

        // O += P @ V  (A = P 16x32 from LDS, B = Vt columns)
        V16 pa;
        {
            const uint4* pp = (const uint4*)&Pw[wid][L][0];
            pa.q[0] = pp[kh];
            pa.q[1] = pp[2 + kh];
        }
#pragma unroll
        for (int d = 0; d < 4; ++d) {
            V16 vb;
            const uint4* vv = (const uint4*)&Vt[d * 16 + L][0];
            vb.q[0] = vv[kh];
            vb.q[1] = vv[2 + kh];
            o[d] = __builtin_amdgcn_wmma_f32_16x16x32_bf16(
                false, pa.v, false, vb.v, (short)0, o[d], false, false);
        }
        __syncthreads();
    }

    // write attn_v as bf16 in [token][h*64 + d] layout
#pragma unroll
    for (int r = 0; r < 8; ++r) {
        float inv = 1.0f / l[r];
        int n_idx = q0 + r + 8 * kh;
        size_t row = ((size_t)b * 2048 + n_idx) * 1024 + h * 64;
#pragma unroll
        for (int d = 0; d < 4; ++d) O[row + d * 16 + L] = f2bf(o[d][r] * inv);
    }
}

// ---------------------------------------------------------------- final LN
__global__ __launch_bounds__(256) void final_ln(const float* __restrict__ acc,
                                                float* __restrict__ out) {
    __shared__ float red[16];
    const int row = blockIdx.x, t = threadIdx.x;
    const float* x = acc + (size_t)row * 1024;
    float v[4], sum = 0.f, sq = 0.f;
#pragma unroll
    for (int i = 0; i < 4; ++i) {
        v[i] = x[t + 256 * i];
        sum += v[i];
        sq  += v[i] * v[i];
    }
#pragma unroll
    for (int o = 16; o; o >>= 1) {
        sum += __shfl_xor(sum, o, 32);
        sq  += __shfl_xor(sq, o, 32);
    }
    const int wid = t >> 5, lane = t & 31;
    if (lane == 0) { red[wid] = sum; red[wid + 8] = sq; }
    __syncthreads();
    float ts = 0.f, tq = 0.f;
#pragma unroll
    for (int i = 0; i < 8; ++i) { ts += red[i]; tq += red[i + 8]; }
    float mu = ts * (1.0f / 1024.0f);
    float rs = rsqrtf(tq * (1.0f / 1024.0f) - mu * mu + 1e-5f);
    float* y = out + (size_t)row * 1024;
#pragma unroll
    for (int i = 0; i < 4; ++i) y[t + 256 * i] = (v[i] - mu) * rs;
}

// ---------------------------------------------------------------- launch
extern "C" void kernel_launch(void* const* d_in, const int* in_sizes, int n_in,
                              void* d_out, int out_size, void* d_ws, size_t ws_size,
                              hipStream_t stream) {
    const float* x    = (const float*)d_in[0];
    const float* Wqkv = (const float*)d_in[1];
    const float* W1   = (const float*)d_in[2];
    const float* b1   = (const float*)d_in[3];
    const float* W2   = (const float*)d_in[4];
    const float* b2   = (const float*)d_in[5];
    const float* Wout = (const float*)d_in[6];
    const float* bout = (const float*)d_in[7];
    float* out = (float*)d_out;

    char* ws = (char*)d_ws;
    unsigned short* XB    = (unsigned short*)(ws);                 // 16 MB  [8192][1024]
    unsigned short* WQKVT = (unsigned short*)(ws + 16777216);      //  6 MB  [3072][1024]
    unsigned short* W1T   = (unsigned short*)(ws + 23068672);      //  8 MB  [4096][1024]
    unsigned short* W2T   = (unsigned short*)(ws + 31457280);      //  8 MB  [1024][4096]
    unsigned short* WOUTT = (unsigned short*)(ws + 39845888);      //  2 MB  [1024][1024]
    float*          QKV   = (float*)(ws + 41943040);               // 96 MB  [8192][3072]
    unsigned short* QB    = (unsigned short*)(ws + 142606336);     // 16 MB  [b,h,n,64]
    unsigned short* KB    = (unsigned short*)(ws + 159383552);     // 16 MB
    unsigned short* VB    = (unsigned short*)(ws + 176160768);     // 16 MB
    unsigned short* AVB   = (unsigned short*)(ws + 192937984);     // 16 MB  [8192][1024]
    unsigned short* H1B   = (unsigned short*)(ws + 209715200);     // 64 MB  [8192][4096]
    float*          ACC   = (float*)(ws + 276824064);              // 32 MB  [8192][1024]

    cvt_f32_bf16<<<dim3(8192 * 1024 / 1024), dim3(256), 0, stream>>>(x, XB, 8192 * 1024);
    cvt_transpose_bf16<<<dim3(3072 / 32, 1024 / 32), dim3(32, 8), 0, stream>>>(Wqkv, WQKVT, 1024, 3072);
    cvt_transpose_bf16<<<dim3(4096 / 32, 1024 / 32), dim3(32, 8), 0, stream>>>(W1,   W1T,   1024, 4096);
    cvt_transpose_bf16<<<dim3(1024 / 32, 4096 / 32), dim3(32, 8), 0, stream>>>(W2,   W2T,   4096, 1024);
    cvt_transpose_bf16<<<dim3(1024 / 32, 1024 / 32), dim3(32, 8), 0, stream>>>(Wout, WOUTT, 1024, 1024);

    // qkv = x @ Wqkv (fp32 out)
    gemm_bf16<0><<<dim3(3072 / BN, 8192 / BM), dim3(256), 0, stream>>>(
        XB, WQKVT, nullptr, QKV, nullptr, 8192, 3072, 1024);

    // QK layernorm + bf16 repack to [b,h,n,64]
    qk_ln<<<dim3(4 * 16 * 2048), dim3(32), 0, stream>>>(QKV, QB, KB, VB);

    // flash attention -> attn_v bf16 [8192,1024]
    attn<<<dim3(64, 32), dim3(128), 0, stream>>>(QB, KB, VB, AVB);

    // ACC = attn_v @ Wout + bout
    gemm_bf16<1><<<dim3(1024 / BN, 8192 / BM), dim3(256), 0, stream>>>(
        AVB, WOUTT, bout, ACC, nullptr, 8192, 1024, 1024);

    // H1 = gelu(x @ W1 + b1) (bf16 out)
    gemm_bf16<3><<<dim3(4096 / BN, 8192 / BM), dim3(256), 0, stream>>>(
        XB, W1T, b1, nullptr, H1B, 8192, 4096, 1024);

    // ACC += H1 @ W2 + b2
    gemm_bf16<2><<<dim3(1024 / BN, 8192 / BM), dim3(256), 0, stream>>>(
        H1B, W2T, b2, ACC, nullptr, 8192, 1024, 4096);

    // out = LN(ACC)
    final_ln<<<dim3(8192), dim3(256), 0, stream>>>(ACC, out);
}
